// Linear3Bit_82927228551442
// MI455X (gfx1250) — compile-verified, matrix operand order checked by reading, not run on hardware
//
#include <hip/hip_runtime.h>

// ---------------------------------------------------------------------------
// 3-bit quantized linear layer for gfx1250 (MI455X), two-pass:
//   pass 1: dequantize weights -> f16 (d_ws), cast x -> f16 (d_ws)
//   pass 2: f16 WMMA GEMM, 256x128 tile, double-buffered LDS, XOR-swizzled
//   fallback (small ws): fused dequant+GEMM (round-1 kernel)
//   out[m][n] = sum_k x[m][k] * dq(w)[n][k] + bias[n]
//   M = 8192, N = OUT = 4096, K = IN = 4096, quant group = 32 along K
// ---------------------------------------------------------------------------

typedef __attribute__((ext_vector_type(16))) _Float16 v16h;
typedef __attribute__((ext_vector_type(8)))  _Float16 v8h;
typedef __attribute__((ext_vector_type(4)))  _Float16 v4h;
typedef __attribute__((ext_vector_type(8)))  float    v8f;

#define IN_DIM   4096
#define OUT_DIM  4096
#define QGROUP   32
#define KGROUPS  (IN_DIM / QGROUP)      // 128 groups per output row
#define NGROUPS  (OUT_DIM * KGROUPS)    // 524288

union PK  { uint4 u; v4h h[2]; };
union V16 { v16h v; uint4 u[2]; v4h q[4]; };

// ======================= pass 1a: weight dequant ===========================
__global__ __launch_bounds__(256)
void dequant_w_kernel(const int* __restrict__ q3,
                      const _Float16* __restrict__ ranges,
                      _Float16* __restrict__ wh)
{
    const int   g     = blockIdx.x * 256 + threadIdx.x;     // group id
    const float wmin  = (float)ranges[2 * g + 0];
    const float wmax  = (float)ranges[2 * g + 1];
    const float scale = (wmax - wmin) * (1.0f / 7.0f);
    const int4* qb  = (const int4*)(q3 + (size_t)g * QGROUP);
    uint4*      dst = (uint4*)(wh + (size_t)g * QGROUP);
#pragma unroll
    for (int l = 0; l < 4; ++l) {
        const int4 q0 = qb[2 * l + 0];
        const int4 q1 = qb[2 * l + 1];
        PK pk;
        pk.h[0][0] = (_Float16)((float)q0.x * scale + wmin);
        pk.h[0][1] = (_Float16)((float)q0.y * scale + wmin);
        pk.h[0][2] = (_Float16)((float)q0.z * scale + wmin);
        pk.h[0][3] = (_Float16)((float)q0.w * scale + wmin);
        pk.h[1][0] = (_Float16)((float)q1.x * scale + wmin);
        pk.h[1][1] = (_Float16)((float)q1.y * scale + wmin);
        pk.h[1][2] = (_Float16)((float)q1.z * scale + wmin);
        pk.h[1][3] = (_Float16)((float)q1.w * scale + wmin);
        dst[l] = pk.u;
    }
}

// ======================= pass 1b: x -> f16 =================================
__global__ __launch_bounds__(256)
void xcast_kernel(const float* __restrict__ x, _Float16* __restrict__ xh)
{
    const size_t t = (size_t)blockIdx.x * 256 + threadIdx.x;  // 8 floats each
    const float4 a = ((const float4*)x)[2 * t + 0];
    const float4 b = ((const float4*)x)[2 * t + 1];
    PK pk;
    pk.h[0][0] = (_Float16)a.x; pk.h[0][1] = (_Float16)a.y;
    pk.h[0][2] = (_Float16)a.z; pk.h[0][3] = (_Float16)a.w;
    pk.h[1][0] = (_Float16)b.x; pk.h[1][1] = (_Float16)b.y;
    pk.h[1][2] = (_Float16)b.z; pk.h[1][3] = (_Float16)b.w;
    ((uint4*)xh)[t] = pk.u;
}

// ======================= pass 2: f16 WMMA GEMM =============================
// Tile 256(M) x 128(N), BK=32, 8 wave32 each owning 64x64 (16 WMMAs/step).
// LDS: rows of 4x16B chunks, physical chunk = logical ^ ((row>>2)&3):
//   -> every ds_*_b128 is 16B aligned and bank-conflict free.
#define GBM 256
#define GBN 128
#define GBK 32
#define NKT (IN_DIM / GBK)   // 128

__global__ __launch_bounds__(256)
void gemm_f16_wmma(const _Float16* __restrict__ xh,
                   const _Float16* __restrict__ wh,
                   const _Float16* __restrict__ bias,
                   float* __restrict__ out)
{
    __shared__ uint4 As[2][GBM * 4];   // 2 x 16 KB
    __shared__ uint4 Bs[2][GBN * 4];   // 2 x  8 KB

    const int tid   = threadIdx.x;
    const int lane  = tid & 31;
    const int wave  = tid >> 5;         // 0..7
    const int wm    = wave >> 1;        // 0..3 : wave row (64 M)
    const int wn    = wave & 1;         // 0..1 : wave col (64 N)
    const int lrow  = lane & 15;
    const int lhalf = lane >> 4;        // K chunk select (layout spec)
    const int m0 = blockIdx.y * GBM;
    const int n0 = blockIdx.x * GBN;

    // ---- cooperative staging maps (fixed per thread) ----
    int aidx[4]; const _Float16* aptr[4];
#pragma unroll
    for (int i = 0; i < 4; ++i) {
        const int f = tid + 256 * i, r = f >> 2, c = f & 3;
        aidx[i] = r * 4 + (c ^ ((r >> 2) & 3));
        aptr[i] = xh + (size_t)(m0 + r) * IN_DIM + c * 8;
    }
    int bidx[2]; const _Float16* bptr[2];
#pragma unroll
    for (int i = 0; i < 2; ++i) {
        const int f = tid + 256 * i, r = f >> 2, c = f & 3;
        bidx[i] = r * 4 + (c ^ ((r >> 2) & 3));
        bptr[i] = wh + (size_t)(n0 + r) * IN_DIM + c * 8;
    }

    // ---- fragment LDS indices (fixed per lane) ----
    int afidx[4][2], bfidx[4][2];
#pragma unroll
    for (int i = 0; i < 4; ++i) {
        const int r = wm * 64 + i * 16 + lrow, s = (r >> 2) & 3;
        afidx[i][0] = r * 4 + (lhalf ^ s);          // K kbase+0..7
        afidx[i][1] = r * 4 + ((lhalf | 2) ^ s);    // K kbase+16..23
    }
#pragma unroll
    for (int j = 0; j < 4; ++j) {
        const int r = wn * 64 + j * 16 + lrow, s = (r >> 2) & 3;
        bfidx[j][0] = r * 4 + (lhalf ^ s);
        bfidx[j][1] = r * 4 + ((lhalf | 2) ^ s);
    }

    // ---- preload tile 0 into buffer 0 ----
    uint4 ra[4], rb[2];
#pragma unroll
    for (int i = 0; i < 4; ++i) ra[i] = *(const uint4*)(aptr[i]);
#pragma unroll
    for (int i = 0; i < 2; ++i) rb[i] = *(const uint4*)(bptr[i]);
#pragma unroll
    for (int i = 0; i < 4; ++i) As[0][aidx[i]] = ra[i];
#pragma unroll
    for (int i = 0; i < 2; ++i) Bs[0][bidx[i]] = rb[i];
    __syncthreads();

    v8f acc[4][4];
    const v8f zero = {};
#pragma unroll
    for (int i = 0; i < 4; ++i)
#pragma unroll
        for (int j = 0; j < 4; ++j) acc[i][j] = zero;

    for (int t = 0; t < NKT; ++t) {
        const int  cur  = t & 1;
        const bool more = (t + 1) < NKT;
        if (more) {             // issue next tile's global loads early
            const int ko = (t + 1) * GBK;
#pragma unroll
            for (int i = 0; i < 4; ++i) ra[i] = *(const uint4*)(aptr[i] + ko);
#pragma unroll
            for (int i = 0; i < 2; ++i) rb[i] = *(const uint4*)(bptr[i] + ko);
        }

        V16 a[4], b[4];
#pragma unroll
        for (int i = 0; i < 4; ++i) {
            a[i].u[0] = As[cur][afidx[i][0]];
            a[i].u[1] = As[cur][afidx[i][1]];
        }
#pragma unroll
        for (int j = 0; j < 4; ++j) {
            b[j].u[0] = Bs[cur][bfidx[j][0]];
            b[j].u[1] = Bs[cur][bfidx[j][1]];
        }
#pragma unroll
        for (int i = 0; i < 4; ++i)
#pragma unroll
            for (int j = 0; j < 4; ++j)
                acc[i][j] = __builtin_amdgcn_wmma_f32_16x16x32_f16(
                    false, a[i].v, false, b[j].v, (short)0, acc[i][j],
                    false, false);

        if (more) {             // stage next tile into the other buffer
#pragma unroll
            for (int i = 0; i < 4; ++i) As[cur ^ 1][aidx[i]] = ra[i];
#pragma unroll
            for (int i = 0; i < 2; ++i) Bs[cur ^ 1][bidx[i]] = rb[i];
        }
        __syncthreads();
    }

    // ---- epilogue: C layout N = lane%16, M = vgpr + 8*(lane/16) ----
#pragma unroll
    for (int j = 0; j < 4; ++j) {
        const int   n  = n0 + wn * 64 + j * 16 + lrow;
        const float bf = (float)bias[n];
#pragma unroll
        for (int i = 0; i < 4; ++i) {
            const int mb = m0 + wm * 64 + i * 16 + lhalf * 8;
#pragma unroll
            for (int e = 0; e < 8; ++e)
                out[(size_t)(mb + e) * OUT_DIM + n] = acc[i][j][e] + bf;
        }
    }
}

// ================= fallback: fused dequant + GEMM (round-1) ================
#define BM   128
#define BN   128
#define BK   32
#define LDA  (BK + 8)

__global__ __launch_bounds__(256)
void linear3bit_fused(const float* __restrict__ x,
                      const int* __restrict__ q3,
                      const _Float16* __restrict__ ranges,
                      const _Float16* __restrict__ bias,
                      float* __restrict__ out)
{
    __shared__ _Float16 As[BM * LDA];
    __shared__ _Float16 Bs[BN * LDA];

    const int tid   = threadIdx.x;
    const int lane  = tid & 31;
    const int wave  = tid >> 5;
    const int wm    = wave >> 2;
    const int wn    = wave & 3;
    const int lrow  = lane & 15;
    const int lhalf = lane >> 4;
    const int kbase = lhalf * 8;
    const int n0 = blockIdx.x * BN;
    const int m0 = blockIdx.y * BM;
    const int wr = tid >> 1;
    const int whf = tid & 1;

    v8f acc[4][2];
    const v8f zero = {};
#pragma unroll
    for (int i = 0; i < 4; ++i)
#pragma unroll
        for (int j = 0; j < 2; ++j) acc[i][j] = zero;

    for (int kt = 0; kt < IN_DIM; kt += BK) {
        {
            const float4* xb = (const float4*)(x + (size_t)m0 * IN_DIM + kt);
#pragma unroll
            for (int i = 0; i < 4; ++i) {
                const int f = tid + 256 * i, r = f >> 3, c4 = f & 7;
                const float4 v = xb[(size_t)r * (IN_DIM / 4) + c4];
                v4h h;
                h[0] = (_Float16)v.x; h[1] = (_Float16)v.y;
                h[2] = (_Float16)v.z; h[3] = (_Float16)v.w;
                *(v4h*)&As[r * LDA + c4 * 4] = h;
            }
        }
        {
            const size_t g = (size_t)(n0 + wr) * KGROUPS + (kt >> 5);
            const float  wmin  = (float)ranges[g * 2 + 0];
            const float  wmax  = (float)ranges[g * 2 + 1];
            const float  scale = (wmax - wmin) * (1.0f / 7.0f);
            const int4*  qb    = (const int4*)(q3 + g * QGROUP) + whf * 4;
            _Float16*    dst   = &Bs[wr * LDA + whf * 16];
#pragma unroll
            for (int l = 0; l < 4; ++l) {
                const int4 qv = qb[l];
                v4h h;
                h[0] = (_Float16)((float)qv.x * scale + wmin);
                h[1] = (_Float16)((float)qv.y * scale + wmin);
                h[2] = (_Float16)((float)qv.z * scale + wmin);
                h[3] = (_Float16)((float)qv.w * scale + wmin);
                *(v4h*)(dst + l * 4) = h;
            }
        }
        __syncthreads();

        V16 a[4], b[2];
#pragma unroll
        for (int i = 0; i < 4; ++i) {
            const _Float16* p = &As[(wm * 64 + i * 16 + lrow) * LDA + kbase];
            a[i].q[0] = *(const v4h*)(p + 0);  a[i].q[1] = *(const v4h*)(p + 4);
            a[i].q[2] = *(const v4h*)(p + 16); a[i].q[3] = *(const v4h*)(p + 20);
        }
#pragma unroll
        for (int j = 0; j < 2; ++j) {
            const _Float16* p = &Bs[(wn * 32 + j * 16 + lrow) * LDA + kbase];
            b[j].q[0] = *(const v4h*)(p + 0);  b[j].q[1] = *(const v4h*)(p + 4);
            b[j].q[2] = *(const v4h*)(p + 16); b[j].q[3] = *(const v4h*)(p + 20);
        }
#pragma unroll
        for (int i = 0; i < 4; ++i)
#pragma unroll
            for (int j = 0; j < 2; ++j)
                acc[i][j] = __builtin_amdgcn_wmma_f32_16x16x32_f16(
                    false, a[i].v, false, b[j].v, (short)0, acc[i][j],
                    false, false);
        __syncthreads();
    }

#pragma unroll
    for (int j = 0; j < 2; ++j) {
        const int   n  = n0 + wn * 32 + j * 16 + lrow;
        const float bf = (float)bias[n];
#pragma unroll
        for (int i = 0; i < 4; ++i) {
            const int mb = m0 + wm * 64 + i * 16 + lhalf * 8;
#pragma unroll
            for (int e = 0; e < 8; ++e)
                out[(size_t)(mb + e) * OUT_DIM + n] = acc[i][j][e] + bf;
        }
    }
}

// ===========================================================================
extern "C" void kernel_launch(void* const* d_in, const int* in_sizes, int n_in,
                              void* d_out, int out_size, void* d_ws, size_t ws_size,
                              hipStream_t stream) {
    const float*    x      = (const float*)d_in[0];      // (4,2048,4096) f32
    const int*      q3     = (const int*)d_in[1];        // (NGROUPS,32) i32
    const _Float16* ranges = (const _Float16*)d_in[2];   // (NGROUPS,2) f16
    const _Float16* bias   = (const _Float16*)d_in[3];   // (4096,) f16
    float*          out    = (float*)d_out;              // (4,2048,4096) f32

    const int Mtotal = in_sizes[0] / IN_DIM;             // 8192

    const size_t wh_bytes = (size_t)OUT_DIM * IN_DIM * sizeof(_Float16); // 32 MiB
    const size_t xh_bytes = (size_t)Mtotal * IN_DIM * sizeof(_Float16);  // 64 MiB

    if (ws_size >= wh_bytes + xh_bytes) {
        _Float16* wh = (_Float16*)d_ws;
        _Float16* xh = (_Float16*)((char*)d_ws + wh_bytes);

        dequant_w_kernel<<<NGROUPS / 256, 256, 0, stream>>>(q3, ranges, wh);
        xcast_kernel<<<(int)(((size_t)Mtotal * IN_DIM / 8) / 256), 256, 0, stream>>>(x, xh);
        dim3 grid(OUT_DIM / GBN, Mtotal / GBM);          // 32 x 32
        gemm_f16_wmma<<<grid, 256, 0, stream>>>(xh, wh, bias, out);
    } else {
        dim3 grid(OUT_DIM / BN, Mtotal / BM);            // 32 x 64
        linear3bit_fused<<<grid, 256, 0, stream>>>(x, q3, ranges, bias, out);
    }
}
